// RNNEncoder_17239998726628
// MI455X (gfx1250) — compile-verified, matrix operand order checked by reading, not run on hardware
//
#include <hip/hip_runtime.h>

// ---------------------------------------------------------------------------
// LSTM encoder for MI455X (gfx1250, wave32, WMMA bf16 16x16x32).
//   B=128, T=1024, IN=130 (use first 128), H=512, G=4H=2048, K=128+512=640
// 8 workgroups x 512 threads (16 waves). WG g owns batch rows [16g,16g+16).
// Per step: gates(16x2048) = A(16x640) x Wcat(640x2048)  via v_wmma_f32_16x16x32_bf16
// Wave w owns hidden cols [32w,32w+32) in all 4 gate quadrants (8 n-tiles).
// ---------------------------------------------------------------------------

typedef __attribute__((ext_vector_type(16))) __bf16 v16bf;
typedef __attribute__((ext_vector_type(8)))  float  v8f;

#define B_      128
#define T_      1024
#define IN_     130
#define H_      512
#define G_      2048            // 4*H
#define K_      640             // 128 + 512
#define KT_     20              // K / 32
#define KPAD    648             // LDS row stride (elems): 1296B, 16B-aligned, bank-conflict-free
#define THREADS 512
#define NWG     8               // B / 16
#define FRAG_DW 256             // dwords per packed B fragment (32 lanes * 32B)

__device__ __forceinline__ unsigned short f2bf(float f) {
  unsigned u = __float_as_uint(f);
  u += 0x7FFFu + ((u >> 16) & 1u);          // round-to-nearest-even
  return (unsigned short)(u >> 16);
}

union FragU { v16bf v; uint4 q[2]; };

__device__ __forceinline__ float sigmoid_f(float x) {
  return 1.0f / (1.0f + __expf(-x));
}
__device__ __forceinline__ float tanh_f(float x) {
  return 1.0f - 2.0f / (__expf(2.0f * x) + 1.0f);
}

// ---------------------------------------------------------------------------
// Pre-pass 1: pack [W_ih | W_hh] (fp32, row-major (2048,128)/(2048,512)) into
// bf16 fragments laid out exactly as the WMMA B-operand per-lane register
// order, so the main loop's B loads are coalesced 1KB bursts.
// fragment f = ((w*8 + q*2 + s)*20 + kt);  inside: [lane][8 dwords].
// Lane n = lane&15 -> column g = q*512 + w*32 + s*16 + n.
// Lane half: lanes 0-15 hold K {0..7,16..23}, lanes 16-31 hold {8..15,24..31}.
// ---------------------------------------------------------------------------
__global__ __launch_bounds__(256)
void prep_w(const float* __restrict__ Wih, const float* __restrict__ Whh,
            unsigned int* __restrict__ wprep) {
  int f    = blockIdx.x;        // 0..2559
  int tid  = threadIdx.x;       // 0..255
  int lane = tid >> 3;          // 0..31
  int e8   = tid & 7;           // dword (2 bf16) within lane
  int kt   = f % KT_;
  int rem  = f / KT_;
  int s    = rem & 1;
  int q    = (rem >> 1) & 3;
  int w    = rem >> 3;
  int n    = lane & 15;
  int g    = q * 512 + w * 32 + s * 16 + n;
  int halfk = (lane >= 16) ? 8 : 0;
  unsigned short h2[2];
#pragma unroll
  for (int i = 0; i < 2; ++i) {
    int e = e8 * 2 + i;
    int k = kt * 32 + ((e < 8) ? (halfk + e) : (16 + halfk + (e - 8)));
    float v = (k < 128) ? Wih[g * 128 + k] : Whh[g * 512 + (k - 128)];
    h2[i] = f2bf(v);
  }
  wprep[(size_t)f * FRAG_DW + lane * 8 + e8] =
      (unsigned)h2[0] | ((unsigned)h2[1] << 16);
}

// Pre-pass 2: bias_sum = b_ih + b_hh
__global__ __launch_bounds__(256)
void prep_bias(const float* __restrict__ bih, const float* __restrict__ bhh,
               float* __restrict__ bias) {
  int g = blockIdx.x * 256 + threadIdx.x;
  if (g < G_) bias[g] = bih[g] + bhh[g];
}

// ---------------------------------------------------------------------------
// Main recurrent kernel.
// ---------------------------------------------------------------------------
__global__ __launch_bounds__(THREADS)
void lstm_kernel(const float* __restrict__ inputs, const int* __restrict__ len,
                 const float* __restrict__ h0, const float* __restrict__ c0,
                 const unsigned int* __restrict__ wprep,
                 const float* __restrict__ bias, float* __restrict__ out) {
  __shared__ unsigned short sA[16 * KPAD];   // A = [x_t (k<128) | h (k>=128)] bf16

  const int tid   = threadIdx.x;
  const int w     = tid >> 5;                // wave 0..15
  const int lane  = tid & 31;
  const int wg    = blockIdx.x;
  const int brow0 = wg * 16;
  const int nlo   = lane & 15;
  const int hi8   = (lane >= 16) ? 8 : 0;
  const int jbase = w * 32;

  // --- init LDS h region from h0 (bf16) ---
  for (int i = tid; i < 16 * H_; i += THREADS) {
    int m = i >> 9, j = i & 511;
    sA[m * KPAD + 128 + j] = f2bf(h0[(size_t)(brow0 + m) * H_ + j]);
  }

  // --- per-tile bias (broadcast over M), per-row lengths, c0, out regs ---
  float biasn[8];
#pragma unroll
  for (int t8 = 0; t8 < 8; ++t8) {
    int q = t8 >> 1, s = t8 & 1;
    biasn[t8] = bias[q * 512 + jbase + s * 16 + nlo];
  }
  int lenv[8];
#pragma unroll
  for (int r = 0; r < 8; ++r) lenv[r] = len[brow0 + r + hi8];

  v8f cacc[2], outh[2];
#pragma unroll
  for (int s = 0; s < 2; ++s) {
    int j = jbase + s * 16 + nlo;
#pragma unroll
    for (int e = 0; e < 8; ++e) {
      cacc[s][e] = c0[(size_t)(brow0 + e + hi8) * H_ + j];
      outh[s][e] = 0.0f;
    }
  }

  // --- stage x(0): 16 rows x 128 cols, 4 elems/thread ---
  {
    int m  = tid >> 5;
    int k0 = (tid & 31) * 4;
    const float* src = inputs + ((size_t)(brow0 + m) * T_ + 0) * IN_ + k0;
    float2 a = *(const float2*)(src);
    float2 b = *(const float2*)(src + 2);
    unsigned* dst = (unsigned*)&sA[m * KPAD + k0];
    dst[0] = (unsigned)f2bf(a.x) | ((unsigned)f2bf(a.y) << 16);
    dst[1] = (unsigned)f2bf(b.x) | ((unsigned)f2bf(b.y) << 16);
  }

  // B-fragment base for this wave/lane (coalesced 32B per lane)
  const unsigned int* wbase = wprep + (size_t)(w * 8) * KT_ * FRAG_DW + lane * 8;
  const unsigned short* arow = &sA[(lane & 15) * KPAD + hi8];

  for (int t = 0; t < T_; ++t) {
    __syncthreads();                         // h(t), x(t) visible in LDS

    // ---- gates = bias + A x Wcat^T ----
    v8f acc[8];
#pragma unroll
    for (int t8 = 0; t8 < 8; ++t8) {
      float bv = biasn[t8];
#pragma unroll
      for (int e = 0; e < 8; ++e) acc[t8][e] = bv;
    }
#pragma unroll 4
    for (int kt = 0; kt < KT_; ++kt) {
      FragU af;
      af.q[0] = *(const uint4*)(arow + kt * 32);        // ds_load_b128
      af.q[1] = *(const uint4*)(arow + kt * 32 + 16);
      const unsigned int* bp = wbase + kt * FRAG_DW;
#pragma unroll
      for (int t8 = 0; t8 < 8; ++t8) {
        FragU bf_;
        const unsigned int* fp = bp + (size_t)t8 * (KT_ * FRAG_DW);
        bf_.q[0] = *(const uint4*)(fp);                 // global_load_b128
        bf_.q[1] = *(const uint4*)(fp + 4);
        acc[t8] = __builtin_amdgcn_wmma_f32_16x16x32_bf16(
            false, af.v, false, bf_.v, (short)0, acc[t8], false, false);
      }
    }

    __syncthreads();                         // all waves done reading LDS A

    // ---- elementwise LSTM cell: tiles {i:0+s, f:2+s, g:4+s, o:6+s} ----
#pragma unroll
    for (int s = 0; s < 2; ++s) {
      int j = jbase + s * 16 + nlo;
#pragma unroll
      for (int e = 0; e < 8; ++e) {
        float iv = sigmoid_f(acc[0 + s][e]);
        float fv = sigmoid_f(acc[2 + s][e]);
        float gv = tanh_f(acc[4 + s][e]);
        float ov = sigmoid_f(acc[6 + s][e]);
        float cv = fv * cacc[s][e] + iv * gv;
        float hv = ov * tanh_f(cv);
        cacc[s][e] = cv;
        int m = e + hi8;
        sA[m * KPAD + 128 + j] = f2bf(hv);   // h for next step
        outh[s][e] = (t + 1 == lenv[e]) ? hv : outh[s][e];
      }
    }

    // ---- stage x(t+1) (safe: all LDS reads done at 2nd barrier) ----
    if (t + 1 < T_) {
      int m  = tid >> 5;
      int k0 = (tid & 31) * 4;
      const float* src = inputs + ((size_t)(brow0 + m) * T_ + (t + 1)) * IN_ + k0;
      float2 a = *(const float2*)(src);
      float2 b = *(const float2*)(src + 2);
      unsigned* dst = (unsigned*)&sA[m * KPAD + k0];
      dst[0] = (unsigned)f2bf(a.x) | ((unsigned)f2bf(a.y) << 16);
      dst[1] = (unsigned)f2bf(b.x) | ((unsigned)f2bf(b.y) << 16);
    }
  }

  // ---- final output: out[b, j] fp32 ----
#pragma unroll
  for (int s = 0; s < 2; ++s) {
    int j = jbase + s * 16 + nlo;
#pragma unroll
    for (int e = 0; e < 8; ++e) {
      out[(size_t)(brow0 + e + hi8) * H_ + j] = outh[s][e];
    }
  }
}

// ---------------------------------------------------------------------------
extern "C" void kernel_launch(void* const* d_in, const int* in_sizes, int n_in,
                              void* d_out, int out_size, void* d_ws, size_t ws_size,
                              hipStream_t stream) {
  (void)in_sizes; (void)n_in; (void)out_size; (void)ws_size;
  const float* inputs = (const float*)d_in[0];
  const int*   len    = (const int*)d_in[1];
  const float* h0     = (const float*)d_in[2];
  const float* c0     = (const float*)d_in[3];
  const float* Wih    = (const float*)d_in[4];
  const float* Whh    = (const float*)d_in[5];
  const float* bih    = (const float*)d_in[6];
  const float* bhh    = (const float*)d_in[7];

  unsigned int* wprep = (unsigned int*)d_ws;                    // 2.56 MB
  float* bias = (float*)((char*)d_ws + (size_t)2560 * 1024);    // +8 KB

  prep_w<<<2560, 256, 0, stream>>>(Wih, Whh, wprep);
  prep_bias<<<G_ / 256, 256, 0, stream>>>(bih, bhh, bias);
  lstm_kernel<<<NWG, THREADS, 0, stream>>>(inputs, len, h0, c0, wprep, bias,
                                           (float*)d_out);
}
